// GQA_14147622273637
// MI455X (gfx1250) — matrix operand order repaired and checked by reading, not compiled, not run on hardware
//
#include <hip/hip_runtime.h>
#include <stdint.h>

typedef _Float16 v16h __attribute__((ext_vector_type(16)));
typedef _Float16 v8h  __attribute__((ext_vector_type(8)));
typedef float    v8f  __attribute__((ext_vector_type(8)));
typedef int      v4i  __attribute__((ext_vector_type(4)));

// pointer-to-AS typedefs (attribute applies to pointee)
typedef __attribute__((address_space(1))) v4i* global_v4i_ptr;
typedef __attribute__((address_space(3))) v4i* lds_v4i_ptr;

namespace cfg {
constexpr int B = 2, T = 2048, C = 2048, NH = 16, NKVH = 4;
constexpr int HS   = C / NH;        // 128
constexpr int HSH  = HS / 2;        // 64
constexpr int KV   = NKVH * HS;     // 512
constexpr int NQKV = C + 2 * KV;    // 3072
constexpr int M    = B * T;         // 4096
constexpr int NREP = NH / NKVH;     // 4
constexpr float QSCALE = 0.08838834764831845f; // 1/sqrt(HS)
}

// ---------------------------------------------------------------------------
// Async global -> LDS staging (CDNA5 GLOBAL_LOAD_ASYNC_TO_LDS_B128, ASYNCcnt)
// Falls back to synchronous copies if the builtin is unavailable.
// ---------------------------------------------------------------------------
#if __has_builtin(__builtin_amdgcn_global_load_async_to_lds_b128)
#define ASYNC_LDS 1
#else
#define ASYNC_LDS 0
#endif

__device__ __forceinline__ void copy_b128_async(void* lds_dst, const void* gsrc) {
#if ASYNC_LDS
    __builtin_amdgcn_global_load_async_to_lds_b128(
        (global_v4i_ptr)(unsigned long long)(uintptr_t)gsrc,
        (lds_v4i_ptr)(unsigned int)(uintptr_t)lds_dst,
        0, 0);
#else
    *(v8h*)lds_dst = *(const v8h*)gsrc;
#endif
}

__device__ __forceinline__ void wait_async_lds() {
#if ASYNC_LDS
#if __has_builtin(__builtin_amdgcn_s_wait_asynccnt)
    __builtin_amdgcn_s_wait_asynccnt(0);
#else
    asm volatile("s_wait_asynccnt 0" ::: "memory");
#endif
#endif
}

// ---------------------------------------------------------------------------
// WMMA fragment loaders (CDNA5 16-bit layouts, wave32; ISA 7.12.2)
// A 16x32 (MxK): lanes 0-15 -> row=lane, K 0..7 & 16..23 ; lanes 16-31 -> K 8..15 & 24..31
// B 32x16 (KxN) from K-major storage [n][k]: lane n=lane&15, K 0..7 & 16..23 (half 0) / 8..15 & 24..31 (half 1)
// C/D 16x16 f32: vgpr r -> M = r + 8*half, N = lane&15
// ---------------------------------------------------------------------------
__device__ __forceinline__ v16h make_v16(const _Float16* plo, const _Float16* phi) {
    v8h lo = *(const v8h*)plo;
    v8h hi = *(const v8h*)phi;
    v16h r;
#pragma unroll
    for (int i = 0; i < 8; ++i) { r[i] = lo[i]; r[i + 8] = hi[i]; }
    return r;
}

__device__ __forceinline__ v16h load_frag_a(const _Float16* base, int ld, int lane) {
    const int hl = lane >> 4, rr = lane & 15;
    const _Float16* p = base + rr * ld;
    return make_v16(p + 8 * hl, p + 16 + 8 * hl);
}

__device__ __forceinline__ v16h load_frag_b_kmajor(const _Float16* base, int ld, int lane) {
    const int hl = lane >> 4, nn = lane & 15;
    const _Float16* p = base + nn * ld;
    return make_v16(p + 8 * hl, p + 16 + 8 * hl);
}

// ---------------------------------------------------------------------------
// f32 -> f16 streaming convert
// ---------------------------------------------------------------------------
__global__ __launch_bounds__(256) void convert_f16_kernel(const float* __restrict__ src,
                                                          _Float16* __restrict__ dst, int n) {
    int i = blockIdx.x * blockDim.x + threadIdx.x;
    const int stride = gridDim.x * blockDim.x;
    for (; i < n; i += stride) dst[i] = (_Float16)src[i];
}

// ---------------------------------------------------------------------------
// Tiled WMMA GEMM:  D[M,N] = A[M,K](f16) * B[K,N](f16) + bias[N]   (f32 out)
// Block 256 thr = 8 waves, tile 128x128, K-step 32. Wave tile 64x32.
// A tile staged via async-to-LDS; B tile transposed through VGPRs.
// ---------------------------------------------------------------------------
__global__ __launch_bounds__(256) void gemm_f16f32_kernel(const _Float16* __restrict__ A,
                                                          const _Float16* __restrict__ Bm,
                                                          const float* __restrict__ bias,
                                                          float* __restrict__ D,
                                                          int Mn, int Nn, int Kn) {
    __shared__ _Float16 As[128 * 32];  // [m][k]
    __shared__ _Float16 Bs[128 * 40];  // [n][k] transposed, padded

    const int tid  = threadIdx.x;
    const int lane = tid & 31, wv = tid >> 5;
    const int hl   = lane >> 4, ln = lane & 15;
    const int m0   = blockIdx.y * 128, n0 = blockIdx.x * 128;
    const int wm   = (wv >> 2) * 64, wn = (wv & 3) * 32;

    v8f zero = {};
    v8f acc[4][2];
#pragma unroll
    for (int i = 0; i < 4; ++i)
#pragma unroll
        for (int j = 0; j < 2; ++j) acc[i][j] = zero;

    const int amr = tid >> 1, akc = (tid & 1) * 16;
    const int bkr = tid >> 3, bnc = (tid & 7) * 16;

    for (int kk = 0; kk < Kn; kk += 32) {
        // stage A tile 128x32 (row-major, contiguous) -> async DMA to LDS
        {
            const _Float16* src = A + (size_t)(m0 + amr) * Kn + kk + akc;
            _Float16* dst = As + amr * 32 + akc;
            copy_b128_async(dst, src);
            copy_b128_async(dst + 8, src + 8);
        }
        // stage B tile 32x128, transpose to [n][k]
        {
            const _Float16* src = Bm + (size_t)(kk + bkr) * Nn + n0 + bnc;
            v8h b0 = *(const v8h*)src;
            v8h b1 = *(const v8h*)(src + 8);
#pragma unroll
            for (int i = 0; i < 8; ++i) {
                Bs[(bnc + i) * 40 + bkr]     = b0[i];
                Bs[(bnc + 8 + i) * 40 + bkr] = b1[i];
            }
        }
        if (kk + 32 < Kn) {
            __builtin_prefetch(A + (size_t)(m0 + amr) * Kn + kk + 32 + akc, 0, 1);
            __builtin_prefetch(Bm + (size_t)(kk + 32 + bkr) * Nn + n0 + bnc, 0, 1);
        }
        wait_async_lds();
        __syncthreads();

        v16h af[4], bf[2];
#pragma unroll
        for (int i = 0; i < 4; ++i) af[i] = load_frag_a(As + (wm + i * 16) * 32, 32, lane);
#pragma unroll
        for (int j = 0; j < 2; ++j) bf[j] = load_frag_b_kmajor(Bs + (wn + j * 16) * 40, 40, lane);
#pragma unroll
        for (int i = 0; i < 4; ++i)
#pragma unroll
            for (int j = 0; j < 2; ++j)
                acc[i][j] = __builtin_amdgcn_wmma_f32_16x16x32_f16(
                    false, af[i], false, bf[j], (short)0, acc[i][j], false, false);
        __syncthreads();
    }

#pragma unroll
    for (int i = 0; i < 4; ++i)
#pragma unroll
        for (int j = 0; j < 2; ++j) {
            const int ncol = n0 + wn + j * 16 + ln;
            const float bb = bias[ncol];
#pragma unroll
            for (int r = 0; r < 8; ++r) {
                const int mrow = m0 + wm + i * 16 + r + 8 * hl;
                D[(size_t)mrow * Nn + ncol] = acc[i][j][r] + bb;
            }
        }
}

// ---------------------------------------------------------------------------
// RoPE Q: rotate pairs, fold 1/sqrt(HS), write f16 (b, h, t, hs)
// ---------------------------------------------------------------------------
__global__ __launch_bounds__(256) void rope_q_kernel(const float* __restrict__ qkv,
                                                     const float* __restrict__ fc,
                                                     const float* __restrict__ fs,
                                                     _Float16* __restrict__ qh) {
    using namespace cfg;
    const int idx = blockIdx.x * blockDim.x + threadIdx.x;
    if (idx >= B * T * NH * HSH) return;
    const int i = idx % HSH;
    const int h = (idx / HSH) % NH;
    const int t = (idx / (HSH * NH)) % T;
    const int b = idx / (HSH * NH * T);
    const float* src = qkv + (size_t)(b * T + t) * NQKV + h * HS + 2 * i;
    const float xr = src[0], xi = src[1];
    const float c = fc[t * HSH + i], s = fs[t * HSH + i];
    _Float16* dst = qh + ((size_t)(b * NH + h) * T + t) * HS + 2 * i;
    dst[0] = (_Float16)((xr * c - xi * s) * QSCALE);
    dst[1] = (_Float16)((xr * s + xi * c) * QSCALE);
}

// ---------------------------------------------------------------------------
// RoPE K: rotate pairs, write fp32 KV-cache output AND f16 copy (b, kvh, t, hs)
// ---------------------------------------------------------------------------
__global__ __launch_bounds__(256) void rope_k_kernel(const float* __restrict__ qkv,
                                                     const float* __restrict__ fc,
                                                     const float* __restrict__ fs,
                                                     float* __restrict__ kout,
                                                     _Float16* __restrict__ kh) {
    using namespace cfg;
    const int idx = blockIdx.x * blockDim.x + threadIdx.x;
    if (idx >= B * T * NKVH * HSH) return;
    const int i = idx % HSH;
    const int h = (idx / HSH) % NKVH;
    const int t = (idx / (HSH * NKVH)) % T;
    const int b = idx / (HSH * NKVH * T);
    const float* src = qkv + (size_t)(b * T + t) * NQKV + C + h * HS + 2 * i;
    const float xr = src[0], xi = src[1];
    const float c = fc[t * HSH + i], s = fs[t * HSH + i];
    const float orr = xr * c - xi * s, oi = xr * s + xi * c;
    const size_t di = ((size_t)(b * NKVH + h) * T + t) * HS + 2 * i;
    kout[di] = orr;  kout[di + 1] = oi;
    kh[di] = (_Float16)orr;  kh[di + 1] = (_Float16)oi;
}

// ---------------------------------------------------------------------------
// V scatter: fp32 cache output + f16 copy (b, kvh, t, hs)
// ---------------------------------------------------------------------------
__global__ __launch_bounds__(256) void copy_v_kernel(const float* __restrict__ qkv,
                                                     float* __restrict__ vout,
                                                     _Float16* __restrict__ vh) {
    using namespace cfg;
    const int idx = blockIdx.x * blockDim.x + threadIdx.x;
    if (idx >= B * T * KV) return;
    const int j = idx % KV;
    const int t = (idx / KV) % T;
    const int b = idx / (KV * T);
    const float val = qkv[(size_t)(b * T + t) * NQKV + C + KV + j];
    const size_t di = ((size_t)(b * NKVH + j / HS) * T + t) * HS + (j % HS);
    vout[di] = val;
    vh[di] = (_Float16)val;
}

// ---------------------------------------------------------------------------
// Flash attention (causal, GQA). Block = 128 thr = 4 waves; 64 query rows/block,
// 16 rows/wave. Online softmax; WMMA for QK^T and P*V.
// K tile staged once per block via async-to-LDS; V tile transposed through VGPRs.
// ---------------------------------------------------------------------------
__global__ __launch_bounds__(128) void attn_kernel(const _Float16* __restrict__ qh,
                                                   const _Float16* __restrict__ kh,
                                                   const _Float16* __restrict__ vh,
                                                   _Float16* __restrict__ yatt) {
    using namespace cfg;
    __shared__ _Float16 Ks[32 * 136];    // [key][hs] K tile (+pad)
    __shared__ _Float16 Vs[HS * 40];     // [hs][key] transposed V tile (32 keys + pad)
    __shared__ _Float16 Ps[4][16 * 40];  // per-wave P tile 16x32 (+pad)

    const int tid  = threadIdx.x;
    const int lane = tid & 31, wv = tid >> 5;
    const int hl   = lane >> 4, ln = lane & 15;
    const int nqt  = T / 64;
    const int qt   = blockIdx.x % nqt;
    const int h    = (blockIdx.x / nqt) % NH;
    const int b    = blockIdx.x / (nqt * NH);
    const int kvh  = h / NREP;
    const int qbase = qt * 64 + wv * 16;

    // Q fragments (already scaled by 1/sqrt(HS)); HS=128 -> 4 K-chunks of 32
    const _Float16* qrow = qh + ((size_t)(b * NH + h) * T + qbase) * HS;
    v16h aq[4];
#pragma unroll
    for (int kk = 0; kk < 4; ++kk) aq[kk] = load_frag_a(qrow + kk * 32, HS, lane);

    v8f zero = {};
    v8f acc[8];
#pragma unroll
    for (int i = 0; i < 8; ++i) acc[i] = zero;
    float mrow[8], lrow[8];
#pragma unroll
    for (int r = 0; r < 8; ++r) { mrow[r] = -1e30f; lrow[r] = 0.0f; }

    const _Float16* kbase = kh + (size_t)(b * NKVH + kvh) * T * HS;
    const _Float16* vbase = vh + (size_t)(b * NKVH + kvh) * T * HS;

    const int skey = tid >> 2;             // staging: key row handled by this thread
    const int shs0 = (tid & 3) * 32;       // staging: hs chunk base

    const int nkb = qt * 2 + 2;  // key blocks of 32 covering causal span of this block
    for (int kb = 0; kb < nkb; ++kb) {
        // stage K tile (32 keys x 128 hs, row-major) -> async DMA to LDS
        {
            const _Float16* srcK = kbase + (size_t)(kb * 32 + skey) * HS + shs0;
            _Float16* dstK = Ks + skey * 136 + shs0;
#pragma unroll
            for (int j = 0; j < 4; ++j) copy_b128_async(dstK + j * 8, srcK + j * 8);
        }
        // stage V tile (32 keys x 128 hs) transposed into LDS [hs][key]
        {
            const _Float16* srcV = vbase + (size_t)(kb * 32 + skey) * HS + shs0;
#pragma unroll
            for (int c4 = 0; c4 < 4; ++c4) {
                v8h vvv = *(const v8h*)(srcV + c4 * 8);
#pragma unroll
                for (int i = 0; i < 8; ++i) Vs[(shs0 + c4 * 8 + i) * 40 + skey] = vvv[i];
            }
        }
        wait_async_lds();
        __syncthreads();

        // scores S = Q * K^T  (2 N-tiles of 16 keys; K^T fragments from LDS)
        v8f s[2];
        s[0] = zero; s[1] = zero;
#pragma unroll
        for (int kk = 0; kk < 4; ++kk)
#pragma unroll
            for (int nt = 0; nt < 2; ++nt) {
                v16h bk = load_frag_b_kmajor(Ks + (nt * 16) * 136 + kk * 32, 136, lane);
                s[nt] = __builtin_amdgcn_wmma_f32_16x16x32_f16(
                    false, aq[kk], false, bk, (short)0, s[nt], false, false);
            }

        // causal mask
#pragma unroll
        for (int nt = 0; nt < 2; ++nt) {
            const int key = kb * 32 + nt * 16 + ln;
#pragma unroll
            for (int r = 0; r < 8; ++r)
                if (key > qbase + r + 8 * hl) s[nt][r] = -1e30f;
        }

        // online softmax (row stats reduced across the 16-lane half)
        float alpha[8];
#pragma unroll
        for (int r = 0; r < 8; ++r) {
            float mb = fmaxf(s[0][r], s[1][r]);
#pragma unroll
            for (int off = 8; off >= 1; off >>= 1) mb = fmaxf(mb, __shfl_xor(mb, off, 32));
            const float mn = fmaxf(mrow[r], mb);
            alpha[r] = __expf(mrow[r] - mn);
            const float p0 = __expf(s[0][r] - mn);
            const float p1 = __expf(s[1][r] - mn);
            s[0][r] = p0; s[1][r] = p1;
            float ps = p0 + p1;
#pragma unroll
            for (int off = 8; off >= 1; off >>= 1) ps += __shfl_xor(ps, off, 32);
            lrow[r] = lrow[r] * alpha[r] + ps;
            mrow[r] = mn;
        }

        // rescale output accumulators
#pragma unroll
        for (int i = 0; i < 8; ++i)
#pragma unroll
            for (int r = 0; r < 8; ++r) acc[i][r] *= alpha[r];

        // P: C-layout -> LDS -> A-layout (f16)
#pragma unroll
        for (int nt = 0; nt < 2; ++nt)
#pragma unroll
            for (int r = 0; r < 8; ++r)
                Ps[wv][(r + 8 * hl) * 40 + nt * 16 + ln] = (_Float16)s[nt][r];

        // O += P * V   (8 N-tiles of hs)
        v16h ap = load_frag_a(&Ps[wv][0], 40, lane);
#pragma unroll
        for (int i = 0; i < 8; ++i) {
            v16h bv = load_frag_b_kmajor(Vs + (i * 16) * 40, 40, lane);
            acc[i] = __builtin_amdgcn_wmma_f32_16x16x32_f16(
                false, ap, false, bv, (short)0, acc[i], false, false);
        }
        __syncthreads();
    }

    // finalize: divide by row sums, write f16 (b, t, h*HS + hs) for projection GEMM
    float inv[8];
#pragma unroll
    for (int r = 0; r < 8; ++r) inv[r] = 1.0f / lrow[r];
#pragma unroll
    for (int i = 0; i < 8; ++i)
#pragma unroll
        for (int r = 0; r < 8; ++r) {
            const int t = qbase + r + 8 * hl;
            yatt[(size_t)(b * T + t) * C + h * HS + i * 16 + ln] =
                (_Float16)(acc[i][r] * inv[r]);
        }
}

// ---------------------------------------------------------------------------
// Launch
// ---------------------------------------------------------------------------
extern "C" void kernel_launch(void* const* d_in, const int* in_sizes, int n_in,
                              void* d_out, int out_size, void* d_ws, size_t ws_size,
                              hipStream_t stream) {
    using namespace cfg;
    (void)in_sizes; (void)n_in; (void)out_size; (void)ws_size;

    const float* x      = (const float*)d_in[0];
    const float* w_qkv  = (const float*)d_in[1];
    const float* b_qkv  = (const float*)d_in[2];
    const float* w_proj = (const float*)d_in[3];
    const float* b_proj = (const float*)d_in[4];
    const float* fc     = (const float*)d_in[5];
    const float* fs     = (const float*)d_in[6];

    float* out_y = (float*)d_out;
    float* out_k = out_y + (size_t)M * C;
    float* out_v = out_k + (size_t)B * NKVH * T * HS;

    char* ws = (char*)d_ws;
    _Float16* xh   = (_Float16*)ws;  ws += sizeof(_Float16) * (size_t)M * C;
    _Float16* wqh  = (_Float16*)ws;  ws += sizeof(_Float16) * (size_t)C * NQKV;
    _Float16* wph  = (_Float16*)ws;  ws += sizeof(_Float16) * (size_t)C * C;
    float*    qkv  = (float*)ws;     ws += sizeof(float)    * (size_t)M * NQKV;
    _Float16* qh   = (_Float16*)ws;  ws += sizeof(_Float16) * (size_t)B * NH * T * HS;
    _Float16* kh   = (_Float16*)ws;  ws += sizeof(_Float16) * (size_t)B * NKVH * T * HS;
    _Float16* vh   = (_Float16*)ws;  ws += sizeof(_Float16) * (size_t)B * NKVH * T * HS;
    _Float16* yatt = (_Float16*)ws;

    convert_f16_kernel<<<2048, 256, 0, stream>>>(x, xh, M * C);
    convert_f16_kernel<<<2048, 256, 0, stream>>>(w_qkv, wqh, C * NQKV);
    convert_f16_kernel<<<2048, 256, 0, stream>>>(w_proj, wph, C * C);

    gemm_f16f32_kernel<<<dim3(NQKV / 128, M / 128), 256, 0, stream>>>(
        xh, wqh, b_qkv, qkv, M, NQKV, C);

    { int tot = B * T * NH * HSH;   rope_q_kernel<<<(tot + 255) / 256, 256, 0, stream>>>(qkv, fc, fs, qh); }
    { int tot = B * T * NKVH * HSH; rope_k_kernel<<<(tot + 255) / 256, 256, 0, stream>>>(qkv, fc, fs, out_k, kh); }
    { int tot = B * T * KV;         copy_v_kernel<<<(tot + 255) / 256, 256, 0, stream>>>(qkv, out_v, vh); }

    attn_kernel<<<B * NH * (T / 64), 128, 0, stream>>>(qh, kh, vh, yatt);

    gemm_f16f32_kernel<<<dim3(C / 128, M / 128), 256, 0, stream>>>(
        yatt, wph, b_proj, out_y, M, C, C);
}
